// VectorQuantizedSampler_59811714564784
// MI455X (gfx1250) — matrix operand order repaired and checked by reading, not compiled, not run on hardware
//
#include <hip/hip_runtime.h>

typedef __attribute__((ext_vector_type(16))) __bf16 v16bf;
typedef __attribute__((ext_vector_type(8)))  __bf16 v8bf;
typedef __attribute__((ext_vector_type(4)))  __bf16 v4bf;
typedef __attribute__((ext_vector_type(8)))  float  v8f;

#define VQ_B 8192
#define VQ_K 8192
#define VQ_D 256
#define LDS_STRIDE 264   // 256 + 8 bf16 pad -> row stride 132 dwords, 4*i banks distinct for 16 lanes

// ---------------------------------------------------------------------------
// Kernel 1: split embeddings f32 -> bf16 hi/lo pair and compute ||e_k||^2.
// One wave per row (8 rows / 256-thread block).
// ---------------------------------------------------------------------------
__global__ __launch_bounds__(256) void vq_prep_kernel(
    const float* __restrict__ emb,
    __bf16* __restrict__ eh, __bf16* __restrict__ el,
    float* __restrict__ esq)
{
    const int wave = threadIdx.x >> 5;
    const int lane = threadIdx.x & 31;
    const int k    = blockIdx.x * 8 + wave;
    const float* row = emb + (size_t)k * VQ_D;

    float ss = 0.0f;
    #pragma unroll
    for (int half = 0; half < 2; ++half) {
        const int col = half * 128 + lane * 4;
        float4 v = *(const float4*)(row + col);
        v4bf h, l;
        h.x = (__bf16)v.x; l.x = (__bf16)(v.x - (float)h.x);
        h.y = (__bf16)v.y; l.y = (__bf16)(v.y - (float)h.y);
        h.z = (__bf16)v.z; l.z = (__bf16)(v.z - (float)h.z);
        h.w = (__bf16)v.w; l.w = (__bf16)(v.w - (float)h.w);
        *(v4bf*)(eh + (size_t)k * VQ_D + col) = h;
        *(v4bf*)(el + (size_t)k * VQ_D + col) = l;
        ss += v.x * v.x + v.y * v.y + v.z * v.z + v.w * v.w;
    }
    #pragma unroll
    for (int m = 16; m >= 1; m >>= 1) ss += __shfl_xor(ss, m, 32);
    if (lane == 0) esq[k] = ss;
}

// ---------------------------------------------------------------------------
// Kernel 2: WMMA cross-product + fused argmin + gather.
// Block = 256 threads (8 waves). Block owns 16 z-rows; each wave scans all K
// in 128-wide chunks (16 columns / wave). Three independent bf16x3 WMMA
// accumulator chains (ISA §7.5) + double-buffered B operands so each B fetch
// is in flight a full 3-WMMA window before its s_wait_loadcnt.
// ---------------------------------------------------------------------------
__global__ __launch_bounds__(256) void vq_argmin_kernel(
    const float*  __restrict__ z,
    const float*  __restrict__ emb,
    const __bf16* __restrict__ eh,
    const __bf16* __restrict__ el,
    const float*  __restrict__ esq,
    float* __restrict__ out)
{
    __shared__ __bf16 zh[16 * LDS_STRIDE];
    __shared__ __bf16 zl[16 * LDS_STRIDE];
    __shared__ float  wval[8][16];
    __shared__ int    widx[8][16];
    __shared__ int    rowIdx[16];

    const int t  = threadIdx.x;
    const int rb = blockIdx.x * 16;

    // --- stage z tile into LDS as split bf16 hi/lo ---
    {
        const int r  = t >> 4;          // 0..15
        const int c0 = (t & 15) * 16;   // 16 floats per thread
        const float* zr = z + (size_t)(rb + r) * VQ_D + c0;
        __bf16* ph = zh + r * LDS_STRIDE + c0;
        __bf16* pl = zl + r * LDS_STRIDE + c0;
        #pragma unroll
        for (int j = 0; j < 16; j += 4) {
            float4 v = *(const float4*)(zr + j);
            v4bf h, l;
            h.x = (__bf16)v.x; l.x = (__bf16)(v.x - (float)h.x);
            h.y = (__bf16)v.y; l.y = (__bf16)(v.y - (float)h.y);
            h.z = (__bf16)v.z; l.z = (__bf16)(v.z - (float)h.z);
            h.w = (__bf16)v.w; l.w = (__bf16)(v.w - (float)h.w);
            *(v4bf*)(ph + j) = h;
            *(v4bf*)(pl + j) = l;
        }
    }
    __syncthreads();

    const int wave = t >> 5;
    const int lane = t & 31;
    const int hi   = lane >> 4;   // half-wave select
    const int n    = lane & 15;   // A: M row / B,C: N column

    // A-matrix rows for this lane live at m == n (ISA 16-bit A layout)
    const __bf16* zhRow = zh + n * LDS_STRIDE;
    const __bf16* zlRow = zl + n * LDS_STRIDE;

    float best[8];
    int   bidx[8];
    #pragma unroll
    for (int i = 0; i < 8; ++i) { best[i] = 3.4e38f; bidx[i] = 0; }

    for (int kc = wave * 16; kc < VQ_K; kc += 128) {
        const __bf16* ehRow = eh + (size_t)(kc + n) * VQ_D + hi * 16;
        const __bf16* elRow = el + (size_t)(kc + n) * VQ_D + hi * 16;
        if (kc + 128 < VQ_K) {
            __builtin_prefetch(eh + (size_t)(kc + 128 + n) * VQ_D, 0, 1);
            __builtin_prefetch(el + (size_t)(kc + 128 + n) * VQ_D, 0, 1);
        }

        // three independent accumulator chains: zh*eh, zh*el, zl*eh
        v8f c0 = {}, c1 = {}, c2 = {};

        // double-buffered B operands
        v16bf Bh = *(const v16bf*)(ehRow);
        v16bf Bl = *(const v16bf*)(elRow);
        #pragma unroll
        for (int d = 0; d < VQ_D; d += 32) {
            // issue next step's B loads before consuming this step's
            v16bf BhN = Bh, BlN = Bl;
            if (d + 32 < VQ_D) {
                BhN = *(const v16bf*)(ehRow + d + 32);
                BlN = *(const v16bf*)(elRow + d + 32);
            }
            // A lane layout: K = {16hi..16hi+15} (contiguous, matches B)
            v8bf  ah0 = *(const v8bf*)(zhRow + d + hi * 16);
            v8bf  ah1 = *(const v8bf*)(zhRow + d + hi * 16 + 8);
            v16bf Ah  = __builtin_shufflevector(ah0, ah1, 0,1,2,3,4,5,6,7,8,9,10,11,12,13,14,15);
            v8bf  al0 = *(const v8bf*)(zlRow + d + hi * 16);
            v8bf  al1 = *(const v8bf*)(zlRow + d + hi * 16 + 8);
            v16bf Al  = __builtin_shufflevector(al0, al1, 0,1,2,3,4,5,6,7,8,9,10,11,12,13,14,15);
            // bf16x3 (zl*el dropped, ~2^-18 relative); independent D/C chains
            c0 = __builtin_amdgcn_wmma_f32_16x16x32_bf16(false, Ah, false, Bh, (short)0, c0, false, false);
            c1 = __builtin_amdgcn_wmma_f32_16x16x32_bf16(false, Ah, false, Bl, (short)0, c1, false, false);
            c2 = __builtin_amdgcn_wmma_f32_16x16x32_bf16(false, Al, false, Bh, (short)0, c2, false, false);
            Bh = BhN;
            Bl = BlN;
        }

        const int   kcol = kc + n;            // C element column for this lane
        const float es   = esq[kcol];
        #pragma unroll
        for (int i = 0; i < 8; ++i) {         // C element i -> row M = i + 8*hi
            const float s = es - 2.0f * (c0[i] + c1[i] + c2[i]);
            if (s < best[i] || (s == best[i] && kcol < bidx[i])) { best[i] = s; bidx[i] = kcol; }
        }
    }

    // reduce across the 16 lanes that share each row (xor masks stay in-half)
    #pragma unroll
    for (int m = 8; m >= 1; m >>= 1) {
        #pragma unroll
        for (int i = 0; i < 8; ++i) {
            const float ov = __shfl_xor(best[i], m, 32);
            const int   oi = __shfl_xor(bidx[i], m, 32);
            if (ov < best[i] || (ov == best[i] && oi < bidx[i])) { best[i] = ov; bidx[i] = oi; }
        }
    }
    if (n == 0) {                 // lane 0 -> rows 0..7, lane 16 -> rows 8..15
        #pragma unroll
        for (int i = 0; i < 8; ++i) { wval[wave][hi * 8 + i] = best[i]; widx[wave][hi * 8 + i] = bidx[i]; }
    }
    __syncthreads();

    if (t < 16) {                 // reduce across the 8 waves
        float bv = wval[0][t];
        int   bi = widx[0][t];
        #pragma unroll
        for (int w = 1; w < 8; ++w) {
            const float ov = wval[w][t];
            const int   oi = widx[w][t];
            if (ov < bv || (ov == bv && oi < bi)) { bv = ov; bi = oi; }
        }
        rowIdx[t] = bi;
    }
    __syncthreads();

    // gather winning f32 embedding rows to output (coalesced float4 copies)
    {
        const int r  = t >> 4;
        const int c0v = (t & 15) * 16;
        const float* src = emb + (size_t)rowIdx[r] * VQ_D + c0v;
        float*       dst = out + (size_t)(rb + r) * VQ_D + c0v;
        #pragma unroll
        for (int j = 0; j < 16; j += 4) *(float4*)(dst + j) = *(const float4*)(src + j);
    }
}

// ---------------------------------------------------------------------------
// Workspace layout: eh (4 MB) | el (4 MB) | esq (32 KB)  -> ~8.03 MB total
// ---------------------------------------------------------------------------
extern "C" void kernel_launch(void* const* d_in, const int* in_sizes, int n_in,
                              void* d_out, int out_size, void* d_ws, size_t ws_size,
                              hipStream_t stream) {
    const float* z   = (const float*)d_in[0];   // [8192, 256] f32
    const float* emb = (const float*)d_in[1];   // [8192, 256] f32
    float* out = (float*)d_out;                 // [8192, 256] f32

    char* ws = (char*)d_ws;
    __bf16* eh  = (__bf16*)ws;
    __bf16* el  = (__bf16*)(ws + (size_t)VQ_K * VQ_D * 2);
    float*  esq = (float*)(ws + (size_t)VQ_K * VQ_D * 4);

    vq_prep_kernel<<<VQ_K / 8, 256, 0, stream>>>(emb, eh, el, esq);
    vq_argmin_kernel<<<VQ_B / 16, 256, 0, stream>>>(z, emb, eh, el, esq, out);
}
